// Transformer_28166395527573
// MI455X (gfx1250) — matrix operand order repaired and checked by reading, not compiled, not run on hardware
//
#include <hip/hip_runtime.h>
#include <hip/hip_bf16.h>

typedef __bf16 bf16;
typedef __attribute__((ext_vector_type(16))) __bf16 v16bf;
typedef __attribute__((ext_vector_type(8)))  __bf16 v8bf;
typedef __attribute__((ext_vector_type(8)))  float  v8f;

#define B_   8
#define S_   512
#define OD_  128
#define E_   1024
#define H_   16
#define L_   8
#define HID_ 4096
#define HD_  64
#define BS_  (B_*S_)   /* 4096 */
#define E3_  (3*E_)    /* 3072 */

// ---------------------------------------------------------------------------
// WMMA helpers (CDNA5 fragment layouts, CDNA5 ISA §7.12.2)
// A-matrix 16x32 bf16: lane(m=lane&15, hf=lane>>4) holds K = hf*8 + {0..7} and
//                      16 + hf*8 + {0..7}  -> two 16B loads, +16 elem apart.
// B-matrix 32x16 bf16: lane(n=lane&15, hf=lane>>4) holds K = hf*8 + {0..7} and
//                      16 + hf*8 + {0..7}  -> two 16B loads, +16 elem apart.
// C/D 16x16 f32: acc[i] is row (i + hf*8), col (lane&15).
// ---------------------------------------------------------------------------
static __device__ inline v16bf ldfragA(const bf16* p) {
  v8bf lo = *(const v8bf*)(p);
  v8bf hi = *(const v8bf*)(p + 16);
  return __builtin_shufflevector(lo, hi, 0,1,2,3,4,5,6,7,8,9,10,11,12,13,14,15);
}
static __device__ inline v16bf ldfragB(const bf16* p) {
  v8bf lo = *(const v8bf*)(p);
  v8bf hi = *(const v8bf*)(p + 16);
  return __builtin_shufflevector(lo, hi, 0,1,2,3,4,5,6,7,8,9,10,11,12,13,14,15);
}
static __device__ inline v8f wmma_bf16(v16bf a, v16bf b, v8f c) {
  return __builtin_amdgcn_wmma_f32_16x16x32_bf16(
      /*neg_a=*/false, a, /*neg_b=*/false, b,
      /*c_mod=*/(short)0, c, /*reuse_a=*/false, /*reuse_b=*/false);
}

// ---------------------------------------------------------------------------
// Generic GEMM:  C[M,N] = A[M,K](bf16,row-major) x W[N,K](fp32,row-major)^T
// fp32 weights converted to bf16 inline during the global->LDS stage so the
// weight stream crosses HBM exactly once (memory-bound workload).
// Block = 256 thr = 8 waves (2x4), tile 64x64, BK=32. M,N,K all multiples of
// 64/64/32 for every call in this model -> no bounds checks.
// ---------------------------------------------------------------------------
__global__ __launch_bounds__(256)
void gemm_bf16w(const bf16* __restrict__ A, int lda,
                const float* __restrict__ W, int ldw,
                const float* __restrict__ bias,
                float* __restrict__ Cf, bf16* __restrict__ Cb, int ldc,
                int K, int relu)
{
  __shared__ bf16 As[64][40];   // +8 pad: avoids ds bank conflicts on frag reads
  __shared__ bf16 Bs[64][40];

  const int tid  = threadIdx.x;
  const int lane = tid & 31;
  const int wave = tid >> 5;
  const int wm = wave >> 2, wn = wave & 3;        // 2 x 4 wave grid
  const int m0 = blockIdx.y * 64;
  const int n0 = blockIdx.x * 64;
  const int lrow = tid >> 2;                      // 0..63
  const int lcol = (tid & 3) * 8;                 // 0,8,16,24
  const int r = lane & 15, hf = lane >> 4;

  v8f acc0 = {}, acc1 = {};
  const bf16*  Ap = A + (size_t)(m0 + lrow) * lda + lcol;
  const float* Wp = W + (size_t)(n0 + lrow) * ldw + lcol;

  for (int k0 = 0; k0 < K; k0 += 32) {
    v8bf av = *(const v8bf*)(Ap + k0);
    float4 w0 = *(const float4*)(Wp + k0);
    float4 w1 = *(const float4*)(Wp + k0 + 4);
    v8bf bv;
    bv[0]=(bf16)w0.x; bv[1]=(bf16)w0.y; bv[2]=(bf16)w0.z; bv[3]=(bf16)w0.w;
    bv[4]=(bf16)w1.x; bv[5]=(bf16)w1.y; bv[6]=(bf16)w1.z; bv[7]=(bf16)w1.w;
    if (k0 + 32 < K) {
      __builtin_prefetch(Ap + k0 + 32, 0, 1);     // global_prefetch_b8
      __builtin_prefetch(Wp + k0 + 32, 0, 1);
    }
    *(v8bf*)&As[lrow][lcol] = av;
    *(v8bf*)&Bs[lrow][lcol] = bv;
    __syncthreads();

    v16bf a0 = ldfragA(&As[wm*32      + r][hf*8]);
    v16bf a1 = ldfragA(&As[wm*32 + 16 + r][hf*8]);
    v16bf b  = ldfragB(&Bs[wn*16      + r][hf*8]);
    acc0 = wmma_bf16(a0, b, acc0);
    acc1 = wmma_bf16(a1, b, acc1);
    __syncthreads();
  }

  const int cn = n0 + wn*16 + r;
  const float bias_v = bias ? bias[cn] : 0.0f;
  for (int i = 0; i < 8; ++i) {
    int cm0 = m0 + wm*32 + i + hf*8;
    int cm1 = cm0 + 16;
    float v0 = acc0[i] + bias_v;
    float v1 = acc1[i] + bias_v;
    if (relu) { v0 = fmaxf(v0, 0.0f); v1 = fmaxf(v1, 0.0f); }
    if (Cf) { Cf[(size_t)cm0*ldc + cn] = v0;        Cf[(size_t)cm1*ldc + cn] = v1; }
    if (Cb) { Cb[(size_t)cm0*ldc + cn] = (bf16)v0;  Cb[(size_t)cm1*ldc + cn] = (bf16)v1; }
  }
}

// ---------------------------------------------------------------------------
// Attention: one wave per (b, h, 16-row Q tile). Scores 16x512 via WMMA,
// wave-shuffle softmax, P·V via WMMA using pre-transposed V.
// qkv: bf16 [B,S,3E], head h occupies cols [h*192, h*192+192): q|k|v of 64.
// vt : bf16 [B,H,HD,S]  (V transposed so B-fragments are contiguous).
// ---------------------------------------------------------------------------
__global__ __launch_bounds__(32)
void attn_kernel(const bf16* __restrict__ qkv, const bf16* __restrict__ vt,
                 bf16* __restrict__ out)
{
  __shared__ float sc[16][512];
  __shared__ bf16  pb[16][512];
  const int lane = threadIdx.x;
  const int r = lane & 15, hf = lane >> 4;
  const int s0 = blockIdx.x * 16;
  const int h  = blockIdx.y;
  const int b  = blockIdx.z;

  const bf16* qbase = qkv + ((size_t)(b*S_ + s0))*E3_ + h*192;
  const bf16* kbase = qkv + ((size_t)(b*S_     ))*E3_ + h*192 + 64;

  // Q fragments for the two K-steps (d = 0..31, 32..63)
  v16bf q0 = ldfragA(qbase + (size_t)r*E3_ + 0  + hf*8);
  v16bf q1 = ldfragA(qbase + (size_t)r*E3_ + 32 + hf*8);

  // S = Q K^T * 1/sqrt(64)
  for (int nt = 0; nt < 32; ++nt) {
    const bf16* kb = kbase + (size_t)(nt*16 + r)*E3_;
    v8f a = {};
    a = wmma_bf16(q0, ldfragB(kb + 0  + hf*8), a);
    a = wmma_bf16(q1, ldfragB(kb + 32 + hf*8), a);
    for (int i = 0; i < 8; ++i)
      sc[i + hf*8][nt*16 + r] = a[i] * 0.125f;
  }
  __syncthreads();

  // softmax over each of the 16 rows (512 wide), lane-parallel
  for (int row = 0; row < 16; ++row) {
    float v[16]; float mx = -3.0e38f;
    for (int j = 0; j < 16; ++j) { v[j] = sc[row][lane + 32*j]; mx = fmaxf(mx, v[j]); }
    for (int off = 16; off >= 1; off >>= 1) mx = fmaxf(mx, __shfl_xor(mx, off, 32));
    float sm = 0.0f;
    for (int j = 0; j < 16; ++j) { v[j] = __expf(v[j] - mx); sm += v[j]; }
    for (int off = 16; off >= 1; off >>= 1) sm += __shfl_xor(sm, off, 32);
    float inv = 1.0f / sm;
    for (int j = 0; j < 16; ++j) pb[row][lane + 32*j] = (bf16)(v[j] * inv);
  }
  __syncthreads();

  // O = P V
  const bf16* vb = vt + ((size_t)(b*H_ + h))*HD_*S_;
  for (int dt = 0; dt < 4; ++dt) {
    v8f o = {};
    for (int ut = 0; ut < 16; ++ut) {
      v16bf pf = ldfragA(&pb[r][ut*32 + hf*8]);
      v16bf vf = ldfragB(vb + (size_t)(dt*16 + r)*S_ + ut*32 + hf*8);
      o = wmma_bf16(pf, vf, o);
    }
    for (int i = 0; i < 8; ++i) {
      int m = i + hf*8;
      out[((size_t)(b*S_ + s0 + m))*E_ + h*HD_ + dt*16 + r] = (bf16)o[i];
    }
  }
}

// ---------------------------------------------------------------------------
// Residual + LayerNorm: z = LN(z + delta)*g + b. Keeps fp32 master, emits bf16
// mirror for the next GEMM. One block per row (E=1024), 256 threads.
// ---------------------------------------------------------------------------
__global__ __launch_bounds__(256)
void ln_kernel(float* __restrict__ zf, const float* __restrict__ delta,
               const float* __restrict__ g, const float* __restrict__ bta,
               bf16* __restrict__ zb)
{
  __shared__ float reds[8], reds2[8];
  const int row = blockIdx.x;
  const int t = threadIdx.x;
  const int lane = t & 31, wave = t >> 5;
  const size_t base = (size_t)row * E_;
  float v[4]; float s = 0.0f, s2 = 0.0f;
  for (int j = 0; j < 4; ++j) {
    int c = t*4 + j;
    float xv = zf[base + c] + delta[base + c];
    v[j] = xv; s += xv; s2 += xv*xv;
  }
  for (int off = 16; off >= 1; off >>= 1) { s += __shfl_xor(s, off, 32); s2 += __shfl_xor(s2, off, 32); }
  if (lane == 0) { reds[wave] = s; reds2[wave] = s2; }
  __syncthreads();
  float ts = 0.0f, ts2 = 0.0f;
  for (int w = 0; w < 8; ++w) { ts += reds[w]; ts2 += reds2[w]; }
  float mean = ts * (1.0f/E_);
  float var  = ts2 * (1.0f/E_) - mean*mean;
  float rstd = rsqrtf(var + 1e-5f);
  for (int j = 0; j < 4; ++j) {
    int c = t*4 + j;
    float y = (v[j] - mean) * rstd * g[c] + bta[c];
    zf[base + c] = y; zb[base + c] = (bf16)y;
  }
}

// ---------------------------------------------------------------------------
// Small setup kernels
// ---------------------------------------------------------------------------
__global__ void gram_kernel(const float* __restrict__ Wobs, float* __restrict__ G) {
  int id = blockIdx.x*256 + threadIdx.x;      // 128*128
  int i = id >> 7, j = id & 127;
  const float* ri = Wobs + (size_t)i*E_;
  const float* rj = Wobs + (size_t)j*E_;
  float acc = 0.0f;
  for (int e = 0; e < E_; ++e) acc += ri[e]*rj[e];
  G[id] = acc;
}

// In-place Gauss-Jordan inversion of SPD 128x128 in LDS (no pivoting needed).
__global__ __launch_bounds__(128)
void ginv_kernel(const float* __restrict__ G, float* __restrict__ Ginv) {
  __shared__ float A[128][129];
  __shared__ float pr[128], pc[128];
  const int j = threadIdx.x;                  // thread owns column j
  for (int i = 0; i < 128; ++i) A[i][j] = G[i*128 + j];
  __syncthreads();
  for (int p = 0; p < 128; ++p) {
    float d = A[p][p];
    __syncthreads();
    float nv = A[p][j] / d;
    pr[j] = (j == p) ? (1.0f / d) : nv;
    pc[j] = (j == p) ? d : A[j][p];
    __syncthreads();
    A[p][j] = pr[j];
    float prj = pr[j];
    for (int i = 0; i < 128; ++i) {
      if (i == p) continue;
      float f = pc[i];
      A[i][j] = (j == p) ? (-f * prj) : (A[i][j] - f * prj);
    }
    __syncthreads();
  }
  for (int i = 0; i < 128; ++i) Ginv[i*128 + j] = A[i][j];
}

__global__ void wobsT_kernel(const float* __restrict__ Wobs, float* __restrict__ WobsT) {
  int idx = blockIdx.x*256 + threadIdx.x;     // E*OD
  int o = idx & 127, e = idx >> 7;
  WobsT[idx] = Wobs[(size_t)o*E_ + e];
}

__global__ void f2bf_kernel(const float* __restrict__ in, bf16* __restrict__ out, int n) {
  int idx = blockIdx.x*256 + threadIdx.x;
  if (idx < n) out[idx] = (bf16)in[idx];
}

// V^T: vt[b][h][d][s] = qkv[b][s][h*192+128+d]
__global__ void vt_kernel(const bf16* __restrict__ qkv, bf16* __restrict__ vt) {
  int idx = blockIdx.x*256 + threadIdx.x;     // B*H*HD*S
  int s = idx & (S_-1);
  int d = (idx >> 9)  & (HD_-1);
  int h = (idx >> 15) & (H_-1);
  int b =  idx >> 19;
  vt[idx] = qkv[((size_t)(b*S_ + s))*E3_ + h*192 + 128 + d];
}

// z += sinusoidal PE; also write bf16 mirror.
__global__ void addpe_kernel(float* __restrict__ zf, bf16* __restrict__ zb) {
  int idx = blockIdx.x*256 + threadIdx.x;     // BS*E
  int e  = idx & (E_-1);
  int bs = idx >> 10;
  int s  = bs & (S_-1);
  int p  = e >> 1;
  float ang = (float)s * __expf((float)(2*p) * (-9.210340371976184f / (float)E_));
  float pe  = (e & 1) ? __cosf(ang) : __sinf(ang);
  float v = zf[idx] + pe;
  zf[idx] = v; zb[idx] = (bf16)v;
}

// ---------------------------------------------------------------------------
extern "C" void kernel_launch(void* const* d_in, const int* in_sizes, int n_in,
                              void* d_out, int out_size, void* d_ws, size_t ws_size,
                              hipStream_t stream)
{
  (void)in_sizes; (void)n_in; (void)out_size; (void)ws_size;
  const float* x    = (const float*)d_in[0];
  const float* Wobs = (const float*)d_in[1];
  const float* Wqkv = (const float*)d_in[2];
  const float* Wo   = (const float*)d_in[3];
  const float* bo   = (const float*)d_in[4];
  const float* ln1g = (const float*)d_in[5];
  const float* ln1b = (const float*)d_in[6];
  const float* W1   = (const float*)d_in[7];
  const float* b1   = (const float*)d_in[8];
  const float* W2   = (const float*)d_in[9];
  const float* b2   = (const float*)d_in[10];
  const float* ln2g = (const float*)d_in[11];
  const float* ln2b = (const float*)d_in[12];

  char* wp = (char*)d_ws;
  auto take = [&](size_t bytes) -> void* {
    void* q = (void*)wp;
    wp += (bytes + 255) & ~(size_t)255;
    return q;
  };
  float* G     = (float*)take((size_t)OD_*OD_*4);
  float* Ginv  = (float*)take((size_t)OD_*OD_*4);
  float* WobsT = (float*)take((size_t)E_*OD_*4);
  bf16*  xbf   = (bf16*) take((size_t)BS_*OD_*2);
  bf16*  ybf   = (bf16*) take((size_t)BS_*OD_*2);
  float* zf    = (float*)take((size_t)BS_*E_*4);
  bf16*  zbf   = (bf16*) take((size_t)BS_*E_*2);
  bf16*  bufA  = (bf16*) take((size_t)BS_*HID_*2);  // qkv (24MB) then ffn hidden (32MB)
  bf16*  qkvbf = bufA;
  bf16*  hbf   = bufA;
  bf16*  vt    = (bf16*) take((size_t)B_*H_*HD_*S_*2);
  bf16*  attnb = (bf16*) take((size_t)BS_*E_*2);
  float* proj  = (float*)take((size_t)BS_*E_*4);

  // ---- embed: pinv via normal equations ----
  gram_kernel <<<(OD_*OD_)/256, 256, 0, stream>>>(Wobs, G);
  ginv_kernel <<<1, 128, 0, stream>>>(G, Ginv);
  wobsT_kernel<<<(E_*OD_)/256, 256, 0, stream>>>(Wobs, WobsT);
  f2bf_kernel <<<(BS_*OD_)/256, 256, 0, stream>>>(x, xbf, BS_*OD_);

  dim3 g_y(OD_/64, BS_/64);   // y = x @ Ginv  (Ginv symmetric -> [N,K] form)
  gemm_bf16w<<<g_y, 256, 0, stream>>>(xbf, OD_, Ginv, OD_, nullptr,
                                      nullptr, ybf, OD_, OD_, 0);
  dim3 g_z0(E_/64, BS_/64);   // z0 = y @ Wobs  (via WobsT as [N,K])
  gemm_bf16w<<<g_z0, 256, 0, stream>>>(ybf, OD_, WobsT, OD_, nullptr,
                                       zf, nullptr, E_, OD_, 0);
  addpe_kernel<<<(BS_*E_)/256, 256, 0, stream>>>(zf, zbf);

  // ---- transformer layers ----
  for (int l = 0; l < L_; ++l) {
    const float* Wqkv_l = Wqkv + (size_t)l*E3_*E_;
    const float* Wo_l   = Wo   + (size_t)l*E_*E_;
    const float* bo_l   = bo   + (size_t)l*E_;
    const float* W1_l   = W1   + (size_t)l*HID_*E_;
    const float* b1_l   = b1   + (size_t)l*HID_;
    const float* W2_l   = W2   + (size_t)l*E_*HID_;
    const float* b2_l   = b2   + (size_t)l*E_;

    dim3 g_qkv(E3_/64, BS_/64);
    gemm_bf16w<<<g_qkv, 256, 0, stream>>>(zbf, E_, Wqkv_l, E_, nullptr,
                                          nullptr, qkvbf, E3_, E_, 0);
    vt_kernel<<<(B_*H_*HD_*S_)/256, 256, 0, stream>>>(qkvbf, vt);

    dim3 g_at(S_/16, H_, B_);
    attn_kernel<<<g_at, 32, 0, stream>>>(qkvbf, vt, attnb);

    dim3 g_o(E_/64, BS_/64);
    gemm_bf16w<<<g_o, 256, 0, stream>>>(attnb, E_, Wo_l, E_, bo_l,
                                        proj, nullptr, E_, E_, 0);
    ln_kernel<<<BS_, 256, 0, stream>>>(zf, proj, ln1g + (size_t)l*E_,
                                       ln1b + (size_t)l*E_, zbf);

    dim3 g_h(HID_/64, BS_/64);
    gemm_bf16w<<<g_h, 256, 0, stream>>>(zbf, E_, W1_l, E_, b1_l,
                                        nullptr, hbf, HID_, E_, 1);
    dim3 g_f(E_/64, BS_/64);
    gemm_bf16w<<<g_f, 256, 0, stream>>>(hbf, HID_, W2_l, HID_, b2_l,
                                        proj, nullptr, E_, HID_, 0);
    ln_kernel<<<BS_, 256, 0, stream>>>(zf, proj, ln2g + (size_t)l*E_,
                                       ln2b + (size_t)l*E_, zbf);
  }

  // ---- output: x_hat = z @ Wobs^T  (Wobs already [OD, E] = [N,K]) ----
  dim3 g_out(OD_/64, BS_/64);
  gemm_bf16w<<<g_out, 256, 0, stream>>>(zbf, E_, Wobs, E_, nullptr,
                                        (float*)d_out, nullptr, OD_, E_, 0);
}
